// AttentionBlock_4294967296478
// MI455X (gfx1250) — compile-verified, matrix operand order checked by reading, not run on hardware
//
#include <hip/hip_runtime.h>

// ---------------- WMMA helpers (gfx1250, wave32) ----------------
typedef _Float16 h16;
typedef __attribute__((ext_vector_type(16))) _Float16 v16h;
typedef __attribute__((ext_vector_type(8)))  float    v8f;

// K position held by VGPR j for lane-half kh (16-bit A/B 16x32 layout, ISA 7.12.2)
__device__ __forceinline__ int kpat(int j, int kh) {
  return ((j >> 2) << 4) + (kh << 3) + ((j & 3) << 1);
}

__device__ __forceinline__ v8f wmma16(v16h a, v16h b, v8f c) {
  return __builtin_amdgcn_wmma_f32_16x16x32_f16(false, a, false, b, (short)0, c, false, false);
}

// 16x32 f16 tile in A layout from a row-contiguous f16 row pointer.
// Per lane this is two contiguous 8-half (16B) runs -> two b128 loads.
__device__ __forceinline__ v16h ldrow_f16(const h16* rowp, int kh) {
  v16h a;
#pragma unroll
  for (int j = 0; j < 8; ++j) {
    int k = kpat(j, kh);
    a[2 * j]     = rowp[k];
    a[2 * j + 1] = rowp[k + 1];
  }
  return a;
}

static constexpr int Bb = 4, Cc = 256, Hh = 128, Ww = 128, HW = 16384, Tt = 64;

// ---------------- Generic GEMM:  Out = act(A @ W^T + bias) (+resid), all f16 in / f16 out ----
// A: M x K row-major f16.  W: N x K row-major f16.  M%64==0, N%64==0.
// Block = 128 threads (4 waves); block tile 64x64; wave tile 16x64; f32 accumulate.
// K / activation / bias / residual are compile-time -> branch-free epilogue, pipelined k-loop.
template <int K, int ACT, bool HAS_BIAS, bool HAS_RESID>
__global__ __launch_bounds__(128) void gemm_wmma_k(
    const h16* __restrict__ A, const h16* __restrict__ W,
    const float* __restrict__ bias, const h16* __restrict__ resid,
    h16* __restrict__ Out, int N)
{
  int wave = threadIdx.x >> 5, lane = threadIdx.x & 31;
  int kh = lane >> 4, nl = lane & 15;
  int row0 = blockIdx.y * 64 + wave * 16;
  int col0 = blockIdx.x * 64;
  v8f acc[4] = {};
  const h16* arow = A + (size_t)(row0 + nl) * K;
  for (int kk = 0; kk < K; kk += 32) {
    __builtin_prefetch(arow + kk + 64, 0, 0);          // global_prefetch_b8, one line ahead
    v16h a = ldrow_f16(arow + kk, kh);
#pragma unroll
    for (int t = 0; t < 4; ++t) {
      v16h bm = ldrow_f16(W + (size_t)(col0 + t * 16 + nl) * K + kk, kh);
      acc[t] = wmma16(a, bm, acc[t]);
    }
  }
#pragma unroll
  for (int t = 0; t < 4; ++t) {
    int n = col0 + t * 16 + nl;
    float bv = HAS_BIAS ? bias[n] : 0.0f;
#pragma unroll
    for (int r = 0; r < 8; ++r) {
      int m = row0 + kh * 8 + r;
      float v = acc[t][r] + bv;
      if (ACT == 1) v = v / (1.0f + __expf(-v));       // SiLU
      if (HAS_RESID) v += (float)resid[(size_t)m * N + n];
      Out[(size_t)m * N + n] = (h16)v;
    }
  }
}

// weight / tensor fp32 -> f16 pack
__global__ __launch_bounds__(256) void pack_f16(const float* __restrict__ in, h16* __restrict__ out, int n) {
  int i = blockIdx.x * 256 + threadIdx.x;
  if (i < n) out[i] = (h16)in[i];
}

// ---------------- Cross attention: q(16384x64) x k,v(64x64) per (b,h), HC=4 ----------------
// grid (B*HC, HW/64), block 128 (4 waves); wave handles 16 q rows.
__global__ __launch_bounds__(128) void cross_sdpa(
    const h16* __restrict__ Q, const h16* __restrict__ Kt,
    const h16* __restrict__ Vt, h16* __restrict__ Out)
{
  __shared__ h16 lds[4][16 * 64];
  int bh = blockIdx.x;
  int b = bh >> 2, h = bh & 3;
  int wave = threadIdx.x >> 5, lane = threadIdx.x & 31;
  int kh = lane >> 4, nl = lane & 15;
  int m0 = blockIdx.y * 64 + wave * 16;
  const h16* qbase = Q  + ((size_t)(b * HW + m0) * Cc) + h * 64;
  const h16* kbase = Kt + ((size_t)(b * Tt) * Cc) + h * 64;
  const h16* vbase = Vt + ((size_t)(b * Tt) * Cc) + h * 64;

  v8f s[4] = {};
#pragma unroll
  for (int kk = 0; kk < 64; kk += 32) {
    v16h a = ldrow_f16(qbase + (size_t)nl * Cc + kk, kh);
#pragma unroll
    for (int t = 0; t < 4; ++t) {
      v16h bm = ldrow_f16(kbase + (size_t)(t * 16 + nl) * Cc + kk, kh);  // K rows = keys
      s[t] = wmma16(a, bm, s[t]);
    }
  }
  // softmax over 64 keys; rows live in 16-lane halves of the D layout
  const float scale = 0.25f;                                 // SHARPEN / sqrt(64)
#pragma unroll
  for (int r = 0; r < 8; ++r) {
    float mx = -1e30f;
#pragma unroll
    for (int t = 0; t < 4; ++t) { s[t][r] *= scale; mx = fmaxf(mx, s[t][r]); }
    for (int o = 1; o < 16; o <<= 1) mx = fmaxf(mx, __shfl_xor(mx, o, 32));
    float sum = 0.0f;
#pragma unroll
    for (int t = 0; t < 4; ++t) { float e = __expf(s[t][r] - mx); s[t][r] = e; sum += e; }
    for (int o = 1; o < 16; o <<= 1) sum += __shfl_xor(sum, o, 32);
    float inv = 1.0f / sum;
#pragma unroll
    for (int t = 0; t < 4; ++t) s[t][r] *= inv;
  }
  // probs D-layout -> LDS row-major [16][64] f16 -> A-layout reload
  h16* pl = lds[wave];
#pragma unroll
  for (int t = 0; t < 4; ++t)
#pragma unroll
    for (int r = 0; r < 8; ++r)
      pl[(kh * 8 + r) * 64 + t * 16 + nl] = (h16)s[t][r];
  __syncthreads();

  v8f o2[4] = {};
#pragma unroll
  for (int kk = 0; kk < 64; kk += 32) {
    v16h a = ldrow_f16(pl + nl * 64 + kk, kh);
#pragma unroll
    for (int t = 0; t < 4; ++t) {
      v16h bm;  // B = V (keys x d): element [k][n] = v[key=k][d=n]  (column loads)
#pragma unroll
      for (int j = 0; j < 8; ++j) {
        int k = kk + kpat(j, kh);
        bm[2 * j]     = vbase[(size_t)k * Cc + t * 16 + nl];
        bm[2 * j + 1] = vbase[(size_t)(k + 1) * Cc + t * 16 + nl];
      }
      o2[t] = wmma16(a, bm, o2[t]);
    }
  }
#pragma unroll
  for (int t = 0; t < 4; ++t)
#pragma unroll
    for (int r = 0; r < 8; ++r)
      Out[((size_t)(b * HW + m0 + kh * 8 + r)) * Cc + h * 64 + t * 16 + nl] = (h16)o2[t][r];
}

// ---------------- Window attention: 8x8 windows, HS=8 heads of 32d ----------------
// grid = B*256 windows * 8 heads, block 128 (4 waves = 4 m-tiles of 16)
__global__ __launch_bounds__(128) void win_sdpa(const h16* __restrict__ QKV, h16* __restrict__ Out)
{
  __shared__ h16 lds[4][16 * 64];
  int id = blockIdx.x;
  int h = id & 7; int win = id >> 3;
  int wx = win & 15, wy = (win >> 4) & 15, b = win >> 8;
  int wave = threadIdx.x >> 5, lane = threadIdx.x & 31;
  int kh = lane >> 4, nl = lane & 15;
  int m0 = wave * 16;
  auto grow = [&](int t) -> size_t {
    return (size_t)b * HW + (size_t)((wy * 8 + (t >> 3)) * Ww + wx * 8 + (t & 7));
  };
  v8f s[4] = {};
  {
    v16h a = ldrow_f16(QKV + grow(m0 + nl) * 768 + h * 32, kh);      // K=32, 1 step
#pragma unroll
    for (int t = 0; t < 4; ++t) {
      v16h bm = ldrow_f16(QKV + grow(t * 16 + nl) * 768 + 256 + h * 32, kh);
      s[t] = wmma16(a, bm, s[t]);
    }
  }
  const float scale = 0.17677669529663687f;                  // 1/sqrt(32)
#pragma unroll
  for (int r = 0; r < 8; ++r) {
    float mx = -1e30f;
#pragma unroll
    for (int t = 0; t < 4; ++t) { s[t][r] *= scale; mx = fmaxf(mx, s[t][r]); }
    for (int o = 1; o < 16; o <<= 1) mx = fmaxf(mx, __shfl_xor(mx, o, 32));
    float sum = 0.0f;
#pragma unroll
    for (int t = 0; t < 4; ++t) { float e = __expf(s[t][r] - mx); s[t][r] = e; sum += e; }
    for (int o = 1; o < 16; o <<= 1) sum += __shfl_xor(sum, o, 32);
    float inv = 1.0f / sum;
#pragma unroll
    for (int t = 0; t < 4; ++t) s[t][r] *= inv;
  }
  h16* pl = lds[wave];
#pragma unroll
  for (int t = 0; t < 4; ++t)
#pragma unroll
    for (int r = 0; r < 8; ++r)
      pl[(kh * 8 + r) * 64 + t * 16 + nl] = (h16)s[t][r];
  __syncthreads();

  v8f o2[2] = {};
#pragma unroll
  for (int kk = 0; kk < 64; kk += 32) {
    v16h a = ldrow_f16(pl + nl * 64 + kk, kh);
#pragma unroll
    for (int t = 0; t < 2; ++t) {
      v16h bm;
#pragma unroll
      for (int j = 0; j < 8; ++j) {
        int k = kk + kpat(j, kh);
        bm[2 * j]     = QKV[grow(k) * 768 + 512 + h * 32 + t * 16 + nl];
        bm[2 * j + 1] = QKV[grow(k + 1) * 768 + 512 + h * 32 + t * 16 + nl];
      }
      o2[t] = wmma16(a, bm, o2[t]);
    }
  }
#pragma unroll
  for (int t = 0; t < 2; ++t)
#pragma unroll
    for (int r = 0; r < 8; ++r)
      Out[grow(m0 + kh * 8 + r) * Cc + h * 32 + t * 16 + nl] = (h16)o2[t][r];
}

// ---------------- Elementwise / normalization kernels ----------------
__global__ __launch_bounds__(256) void gn_partial(const float* __restrict__ x, float* __restrict__ mv) {
  int gi = blockIdx.y;  // b*4+g
  const float* base = x + (size_t)gi * (64 * HW);
  size_t e0 = (size_t)blockIdx.x * 4096 + threadIdx.x;
  float s = 0.0f, s2 = 0.0f;
#pragma unroll
  for (int i = 0; i < 16; ++i) { float v = base[e0 + (size_t)i * 256]; s += v; s2 += v * v; }
  for (int o = 16; o > 0; o >>= 1) { s += __shfl_down(s, o, 32); s2 += __shfl_down(s2, o, 32); }
  __shared__ float smA[8], smB[8];
  int wv = threadIdx.x >> 5;
  if ((threadIdx.x & 31) == 0) { smA[wv] = s; smB[wv] = s2; }
  __syncthreads();
  if (threadIdx.x == 0) {
    float a = 0, bq = 0;
    for (int i = 0; i < 8; ++i) { a += smA[i]; bq += smB[i]; }
    atomicAdd(&mv[gi * 2], a);
    atomicAdd(&mv[gi * 2 + 1], bq);
  }
}

__global__ __launch_bounds__(256) void gn_apply(const float* __restrict__ x, const float* __restrict__ mv,
                                                h16* __restrict__ norm) {
  size_t idx = (size_t)blockIdx.x * 256 + threadIdx.x;   // NCHW index
  int b = (int)(idx >> 22);
  int c = (int)((idx >> 14) & 255);
  int p = (int)(idx & 16383);
  int g = c >> 6;
  const float cnt = 64.0f * 16384.0f;
  float mean = mv[(b * 4 + g) * 2] / cnt;
  float var  = mv[(b * 4 + g) * 2 + 1] / cnt - mean * mean;
  float v = (x[idx] - mean) * rsqrtf(var + 1e-5f);
  norm[((size_t)(b * HW + p)) * Cc + c] = (h16)v;          // NHWC out
}

// small dense: out[b][n] = style[b] . w[n] + bias[n]  (K=512)
__global__ void gp_kern(const float* __restrict__ style, const float* __restrict__ w,
                        const float* __restrict__ bv, float* __restrict__ outp, int N) {
  int b = blockIdx.x, n = threadIdx.x;
  const float* sp = style + (size_t)b * 512;
  const float* wr = w + (size_t)n * 512;
  float acc = bv[n];
  for (int k = 0; k < 512; ++k) acc += sp[k] * wr[k];
  outp[(size_t)b * N + n] = acc;
}

// LN over C=256 -> f16 out. mode 0: A[row%64]+Add[row/64]; 1: A[row]+Add[row%HW]; 2: A[row]
template <typename TA, typename TB>
__global__ __launch_bounds__(256) void ln_rows(const TA* __restrict__ A, const TB* __restrict__ Add,
                                               const float* __restrict__ g, const float* __restrict__ be,
                                               h16* __restrict__ Out, int mode) {
  __shared__ float sm[8], sm2[8];
  int row = blockIdx.x, c = threadIdx.x;
  float v;
  if (mode == 0)      v = (float)A[(row & 63) * Cc + c] + (float)Add[(row >> 6) * Cc + c];
  else if (mode == 1) v = (float)A[(size_t)row * Cc + c] + (float)Add[(size_t)(row & (HW - 1)) * Cc + c];
  else                v = (float)A[(size_t)row * Cc + c];
  float s = v, s2 = v * v;
  for (int o = 16; o > 0; o >>= 1) { s += __shfl_down(s, o, 32); s2 += __shfl_down(s2, o, 32); }
  int wv = threadIdx.x >> 5;
  if ((threadIdx.x & 31) == 0) { sm[wv] = s; sm2[wv] = s2; }
  __syncthreads();
  if (threadIdx.x == 0) {
    float a = 0, bq = 0;
    for (int i = 0; i < 8; ++i) { a += sm[i]; bq += sm2[i]; }
    sm[0] = a; sm2[0] = bq;
  }
  __syncthreads();
  float mean = sm[0] * (1.0f / 256.0f);
  float var  = sm2[0] * (1.0f / 256.0f) - mean * mean;
  Out[(size_t)row * Cc + c] = (h16)((v - mean) * rsqrtf(var + 1e-5f) * g[c] + be[c]);
}

__global__ __launch_bounds__(256) void pos_h1(const float* __restrict__ w1, const float* __restrict__ b1,
                                              h16* __restrict__ h1) {
  int p = blockIdx.x, c = threadIdx.x;
  int y = p >> 7, xq = p & 127;
  float gxv = -1.0f + 2.0f * xq / 127.0f;
  float gyv = -1.0f + 2.0f * y  / 127.0f;
  float z = gxv * w1[c * 2] + gyv * w1[c * 2 + 1] + b1[c];
  h1[(size_t)p * Cc + c] = (h16)(z / (1.0f + __expf(-z)));  // SiLU
}

__global__ __launch_bounds__(256) void hcomb(const float* __restrict__ x, const h16* __restrict__ norm,
                                             const h16* __restrict__ sc, const float* __restrict__ gp,
                                             const float* __restrict__ gamma, h16* __restrict__ hbuf) {
  size_t idx = (size_t)blockIdx.x * 256 + threadIdx.x;     // NHWC index
  int c = (int)(idx & 255);
  size_t bp = idx >> 8;
  int b = (int)(bp >> 14);
  int p = (int)(bp & 16383);
  float scale = gp[b * 512 + c], shift = gp[b * 512 + 256 + c];
  float v = x[((size_t)b * 256 + c) * HW + p] + (float)norm[idx] * scale + shift + (float)sc[idx] * gamma[c];
  hbuf[idx] = (h16)v;
}

__global__ __launch_bounds__(256) void rollk(const h16* __restrict__ h, h16* __restrict__ hr) {
  size_t idx = (size_t)blockIdx.x * 256 + threadIdx.x;     // NHWC (rolled space)
  int c = (int)(idx & 255);
  size_t bp = idx >> 8;
  int b = (int)(bp >> 14);
  int p = (int)(bp & 16383);
  int y = p >> 7, xq = p & 127;
  int pr = (((y + 4) & 127) << 7) | ((xq + 4) & 127);
  hr[idx] = h[((size_t)(b * HW + pr)) * Cc + c];
}

__global__ __launch_bounds__(256) void finalk(const h16* __restrict__ h, const h16* __restrict__ sp,
                                              float* __restrict__ out) {
  size_t idx = (size_t)blockIdx.x * 256 + threadIdx.x;     // NCHW output index
  int p = (int)(idx & 16383);
  int c = (int)((idx >> 14) & 255);
  int b = (int)(idx >> 22);
  int y = p >> 7, xq = p & 127;
  int rr = (((y + 124) & 127) << 7) | ((xq + 124) & 127);  // roll back by +4
  out[idx] = (float)h[((size_t)(b * HW + p)) * Cc + c] + (float)sp[((size_t)(b * HW + rr)) * Cc + c];
}

// ---------------- launch ----------------
extern "C" void kernel_launch(void* const* d_in, const int* in_sizes, int n_in,
                              void* d_out, int out_size, void* d_ws, size_t ws_size,
                              hipStream_t stream) {
  (void)in_sizes; (void)n_in; (void)out_size; (void)ws_size;
  const float* x          = (const float*)d_in[0];
  const float* style      = (const float*)d_in[1];
  const float* qkv_w      = (const float*)d_in[2];
  const float* spat_w     = (const float*)d_in[3];
  const float* gproj_w    = (const float*)d_in[4];
  const float* gproj_b    = (const float*)d_in[5];
  const float* tokens     = (const float*)d_in[6];
  const float* sproj_w    = (const float*)d_in[7];
  const float* sproj_b    = (const float*)d_in[8];
  const float* pos_w1     = (const float*)d_in[9];
  const float* pos_b1     = (const float*)d_in[10];
  const float* pos_w2     = (const float*)d_in[11];
  const float* pos_b2     = (const float*)d_in[12];
  const float* q_w = (const float*)d_in[13]; const float* q_b = (const float*)d_in[14];
  const float* k_w = (const float*)d_in[15]; const float* k_b = (const float*)d_in[16];
  const float* v_w = (const float*)d_in[17]; const float* v_b = (const float*)d_in[18];
  const float* o_w = (const float*)d_in[19]; const float* o_b = (const float*)d_in[20];
  const float* ffn_w1 = (const float*)d_in[21]; const float* ffn_b1 = (const float*)d_in[22];
  const float* ffn_w2 = (const float*)d_in[23]; const float* ffn_b2 = (const float*)d_in[24];
  const float* tn_g = (const float*)d_in[25]; const float* tn_b = (const float*)d_in[26];
  const float* qn_g = (const float*)d_in[27]; const float* qn_b = (const float*)d_in[28];
  const float* fn_g = (const float*)d_in[29]; const float* fn_b = (const float*)d_in[30];
  const float* gamma = (const float*)d_in[31];

  const size_t NT = (size_t)Bb * HW * Cc;                  // 16,777,216 elems
  // ---- f16 region ----
  h16* hw0 = (h16*)d_ws;
  h16* buf0 = hw0;                                          // NT   (norm / hr / sp)
  h16* buf1 = hw0 + NT;                                     // NT
  h16* buf2 = hw0 + 2 * NT;                                 // NT
  h16* buf3 = hw0 + 3 * NT;                                 // 3*NT (ffh then qkv)
  h16* posb = hw0 + 6 * NT;                                 // HW*C
  h16* h1b  = posb + (size_t)HW * Cc;                       // HW*C
  h16* stln = h1b + (size_t)HW * Cc;                        // 4*64*256
  h16* kbuf = stln + 65536;
  h16* vbuf = kbuf + 65536;
  // f16 packed weights
  h16* wq    = vbuf + 65536;        // 65536
  h16* wk    = wq + 65536;
  h16* wv    = wk + 65536;
  h16* wo    = wv + 65536;
  h16* wf1   = wo + 65536;          // 131072
  h16* wf2   = wf1 + 131072;        // 131072
  h16* wqkv  = wf2 + 131072;        // 196608
  h16* wspat = wqkv + 196608;       // 65536
  h16* wpos2 = wspat + 65536;       // 65536
  // ---- f32 region (2-byte counts above are all even -> 4B aligned) ----
  float* fr   = (float*)(wpos2 + 65536);
  float* gpB  = fr;                 // 4*512
  float* sdot = gpB + 2048;         // 4*256
  float* mv   = sdot + 1024;        // 32

  // pack weights fp32 -> f16 (tiny, one-time per call)
  pack_f16<<<256, 256, 0, stream>>>(q_w,     wq,    65536);
  pack_f16<<<256, 256, 0, stream>>>(k_w,     wk,    65536);
  pack_f16<<<256, 256, 0, stream>>>(v_w,     wv,    65536);
  pack_f16<<<256, 256, 0, stream>>>(o_w,     wo,    65536);
  pack_f16<<<512, 256, 0, stream>>>(ffn_w1,  wf1,   131072);
  pack_f16<<<512, 256, 0, stream>>>(ffn_w2,  wf2,   131072);
  pack_f16<<<768, 256, 0, stream>>>(qkv_w,   wqkv,  196608);
  pack_f16<<<256, 256, 0, stream>>>(spat_w,  wspat, 65536);
  pack_f16<<<256, 256, 0, stream>>>(pos_w2,  wpos2, 65536);

  hipMemsetAsync(mv, 0, 32 * sizeof(float), stream);
  gn_partial<<<dim3(256, 16), 256, 0, stream>>>(x, mv);
  gn_apply<<<65536, 256, 0, stream>>>(x, mv, buf0);                         // norm NHWC f16
  gp_kern<<<4, 512, 0, stream>>>(style, gproj_w, gproj_b, gpB, 512);        // scale/shift
  gp_kern<<<4, 256, 0, stream>>>(style, sproj_w, sproj_b, sdot, 256);
  ln_rows<float, float><<<256, 256, 0, stream>>>(tokens, sdot, tn_g, tn_b, stln, 0);
  gemm_wmma_k<256, 0, true,  false><<<dim3(4, 4),    128, 0, stream>>>(stln, wk, k_b, nullptr, kbuf, 256);
  gemm_wmma_k<256, 0, true,  false><<<dim3(4, 4),    128, 0, stream>>>(stln, wv, v_b, nullptr, vbuf, 256);
  pos_h1<<<16384, 256, 0, stream>>>(pos_w1, pos_b1, h1b);
  gemm_wmma_k<256, 0, true,  false><<<dim3(4, 256),  128, 0, stream>>>(h1b, wpos2, pos_b2, nullptr, posb, 256);
  ln_rows<h16, h16><<<65536, 256, 0, stream>>>(buf0, posb, qn_g, qn_b, buf1, 1);     // q_in
  gemm_wmma_k<256, 0, true,  false><<<dim3(4, 1024), 128, 0, stream>>>(buf1, wq, q_b, nullptr, buf2, 256);
  cross_sdpa<<<dim3(16, 256), 128, 0, stream>>>(buf2, kbuf, vbuf, buf1);             // attn out
  gemm_wmma_k<256, 0, true,  false><<<dim3(4, 1024), 128, 0, stream>>>(buf1, wo, o_b, nullptr, buf2, 256);
  ln_rows<h16, h16><<<65536, 256, 0, stream>>>(buf2, nullptr, fn_g, fn_b, buf1, 2);
  gemm_wmma_k<256, 1, true,  false><<<dim3(8, 1024), 128, 0, stream>>>(buf1, wf1, ffn_b1, nullptr, buf3, 512);
  gemm_wmma_k<512, 0, true,  true ><<<dim3(4, 1024), 128, 0, stream>>>(buf3, wf2, ffn_b2, buf2, buf1, 256);
  hcomb<<<65536, 256, 0, stream>>>(x, buf0, buf1, gpB, gamma, buf2);                 // h NHWC f16
  rollk<<<65536, 256, 0, stream>>>(buf2, buf0);                                      // h rolled
  gemm_wmma_k<256, 0, false, false><<<dim3(12, 1024), 128, 0, stream>>>(buf0, wqkv, nullptr, nullptr, buf3, 768);
  win_sdpa<<<8192, 128, 0, stream>>>(buf3, buf1);                                    // window attn
  gemm_wmma_k<256, 0, false, false><<<dim3(4, 1024), 128, 0, stream>>>(buf1, wspat, nullptr, nullptr, buf0, 256);
  finalk<<<65536, 256, 0, stream>>>(buf2, buf0, (float*)d_out);
}
